// QLinear_47923245089349
// MI455X (gfx1250) — matrix operand order repaired and checked by reading, not compiled, not run on hardware
//
#include <hip/hip_runtime.h>

typedef __attribute__((ext_vector_type(16))) _Float16 v16h;
typedef __attribute__((ext_vector_type(8)))  _Float16 v8h;
typedef __attribute__((ext_vector_type(8)))  float    v8f;

#define KDIM 4096        // R: reduction length
#define NDIM 4096        // OUT: columns of out
#define BM   128         // block tile M (rows of x)
#define BN   256         // block tile N (rows of weight)
#define BK   32          // K per WMMA step
#define LSTRIDE 40       // halves per LDS row (32 data + 8 pad), keeps 16B alignment

union FragU { v16h v; v8h h[2]; };

__global__ __launch_bounds__(256)
void qlinear_wmma_f16(const float* __restrict__ x,
                      const int*   __restrict__ w,
                      const int*   __restrict__ bias,
                      float*       __restrict__ out)
{
    __shared__ _Float16 sA[2][BM * LSTRIDE];   // x tile, f16, [m][k]
    __shared__ _Float16 sB[2][BN * LSTRIDE];   // w tile, f16, [n][k]

    const int t    = threadIdx.x;
    const int wave = t >> 5;
    const int lane = t & 31;
    const int llo  = lane & 15;     // row-within-16x16-tile selector
    const int lhi  = lane >> 4;     // K-half selector

    const int m0 = blockIdx.y * BM;
    const int n0 = blockIdx.x * BN;

    // x loader: each thread owns 16 consecutive k of row (t>>1)   (128 rows x 32 k)
    const int arow = t >> 1;
    const int aseg = (t & 1) * 16;
    const float* xg = x + (size_t)(m0 + arow) * KDIM + aseg;
    // w loader: each thread owns the full 32-k slice of row t      (256 rows x 32 k)
    const int* wg = w + (size_t)(n0 + t) * KDIM;

    // wave sub-tile: 64 (M) x 64 (N); waves in 2x4 grid
    const int wm = (wave >> 2) * 64;    // 0 or 64
    const int wn = (wave & 3)  * 64;    // 0,64,128,192

    v8f acc[4][4] = {};                 // 4 M-tiles x 4 N-tiles, f32 accum (128 VGPRs)

    float4 xv[4];                       // 16 f32 staging
    int4   wv[8];                       // 32 i32 staging

    auto fetch = [&](int k0) {
        const float4* xp = (const float4*)(xg + k0);
        const int4*   wp = (const int4*)(wg + k0);
#pragma unroll
        for (int i = 0; i < 4; ++i) xv[i] = xp[i];
#pragma unroll
        for (int i = 0; i < 8; ++i) wv[i] = wp[i];
    };

    auto stage = [&](int buf) {
        union { v8h v[2]; _Float16 h[16]; } ca;
        union { v8h v[4]; _Float16 h[32]; } cb;
#pragma unroll
        for (int i = 0; i < 4; ++i) {
            ca.h[4*i+0] = (_Float16)xv[i].x;  ca.h[4*i+1] = (_Float16)xv[i].y;
            ca.h[4*i+2] = (_Float16)xv[i].z;  ca.h[4*i+3] = (_Float16)xv[i].w;
        }
#pragma unroll
        for (int i = 0; i < 8; ++i) {
            cb.h[4*i+0] = (_Float16)(float)wv[i].x;  cb.h[4*i+1] = (_Float16)(float)wv[i].y;
            cb.h[4*i+2] = (_Float16)(float)wv[i].z;  cb.h[4*i+3] = (_Float16)(float)wv[i].w;
        }
        v8h* pa = (v8h*)&sA[buf][arow * LSTRIDE + aseg];
        pa[0] = ca.v[0]; pa[1] = ca.v[1];
        v8h* pb = (v8h*)&sB[buf][t * LSTRIDE];
#pragma unroll
        for (int i = 0; i < 4; ++i) pb[i] = cb.v[i];
    };

    // prologue: k-tile 0 -> LDS buffer 0
    fetch(0);
    stage(0);
    __syncthreads();

    constexpr int KT = KDIM / BK;       // 128 k-steps
    for (int kt = 0; kt < KT; ++kt) {
        const int buf = kt & 1;

        // issue next tile's global loads before compute so VMEM overlaps WMMA
        if (kt + 1 < KT) fetch((kt + 1) * BK);
        if (kt + 2 < KT) {
            __builtin_prefetch(xg + (kt + 2) * BK, 0, 0);   // global_prefetch_b8
            __builtin_prefetch(wg + (kt + 2) * BK, 0, 0);
        }

        // B fragments held for the whole step: lane -> row wn+nt*16+llo, k = 16*lhi..+15
        FragU bf[4];
#pragma unroll
        for (int nt = 0; nt < 4; ++nt) {
            const _Float16* p = &sB[buf][(wn + nt * 16 + llo) * LSTRIDE + 16 * lhi];
            bf[nt].h[0] = *(const v8h*)p;          // ds_load_b128
            bf[nt].h[1] = *(const v8h*)(p + 8);
        }

        // A fragments loaded just-in-time to cap register pressure
#pragma unroll
        for (int mt = 0; mt < 4; ++mt) {
            FragU af;
            const _Float16* p = &sA[buf][(wm + mt * 16 + llo) * LSTRIDE + 16 * lhi];
            af.h[0] = *(const v8h*)p;              // k = 16*lhi .. +7
            af.h[1] = *(const v8h*)(p + 8);        // k = 16*lhi+8 .. +7
#pragma unroll
            for (int nt = 0; nt < 4; ++nt)
                acc[mt][nt] = __builtin_amdgcn_wmma_f32_16x16x32_f16(
                    false, af.v, false, bf[nt].v,
                    (short)0, acc[mt][nt], false, false);
        }

        // convert + stage next tile into the other buffer
        if (kt + 1 < KT) stage(buf ^ 1);
        __syncthreads();
    }

    // epilogue: out[i][j] = acc/128 + bias[j]
    // C/D layout: lane = (n mod 16) + 16*(m-half); VGPR v holds row m = v + 8*lhi
    const float inv = 1.0f / 128.0f;
#pragma unroll
    for (int mt = 0; mt < 4; ++mt) {
#pragma unroll
        for (int nt = 0; nt < 4; ++nt) {
            const int col   = n0 + wn + nt * 16 + llo;
            const float bj  = (float)bias[col];
            const int rbase = m0 + wm + mt * 16 + 8 * lhi;
            float* op = out + (size_t)rbase * NDIM + col;
#pragma unroll
            for (int v = 0; v < 8; ++v)
                op[(size_t)v * NDIM] = acc[mt][nt][v] * inv + bj;
        }
    }
}

extern "C" void kernel_launch(void* const* d_in, const int* in_sizes, int n_in,
                              void* d_out, int out_size, void* d_ws, size_t ws_size,
                              hipStream_t stream) {
    const float* x    = (const float*)d_in[0];   // (4096, 4096) f32
    const int*   wgt  = (const int*)d_in[1];     // (4096, 4096) i32
    const int*   bias = (const int*)d_in[2];     // (4096,)      i32
    float*       out  = (float*)d_out;           // (4096, 4096) f32

    dim3 grid(NDIM / BN, 4096 / BM);             // (16, 32) blocks
    qlinear_wmma_f16<<<grid, 256, 0, stream>>>(x, wgt, bias, out);
}
